// SimComE_30932354466028
// MI455X (gfx1250) — compile-verified, matrix operand order checked by reading, not run on hardware
//
#include <hip/hip_runtime.h>

// ---------------------------------------------------------------------------
// GIN + virtual-node GNN for MI455X (gfx1250, wave32, WMMA, TDM).
// Needs ~700 MB of workspace (h, ea_cat, agg/t2, t1 padded to H=320 + weights).
// ---------------------------------------------------------------------------

#define NN 100000
#define EE 200000
#define GG 4096
#define HH 300
#define HP 320          // padded hidden (20 x 16 tiles, 10 x 32 K-chunks)
#define LL 4
#define F1N 12
#define F2N 6

#define TM 64           // block tile rows
#define TN 64           // block tile cols
#define KC 160          // K chunk of A staged in LDS (B panel fully resident)

typedef __attribute__((ext_vector_type(16))) __bf16 v16bf;
typedef __attribute__((ext_vector_type(8)))  float  v8f;
typedef __attribute__((ext_vector_type(4)))  unsigned int v4u;
typedef __attribute__((ext_vector_type(8)))  int    v8i;
typedef __attribute__((ext_vector_type(4)))  int    v4i;

#if defined(__HIP_DEVICE_COMPILE__) && defined(__gfx1250__) && \
    __has_builtin(__builtin_amdgcn_tensor_load_to_lds)
#define HAVE_TDM 1
#else
#define HAVE_TDM 0
#endif

enum { A_PLAIN = 0, A_PRE = 1, A_BNRELU = 2 };

// ---------------------------------------------------------------------------
// WMMA GEMM: Out[M x HP] = transform(A)[M x HP(bf16)] @ Bt^T + bias (+ fused
// small-K edge-feature term) and optional per-column sum/sumsq for BN.
// Bt is the bf16 weight stored transposed+padded: Bt[n][k], n,k in [0,HP).
// The 40 KB B panel (rows col0..col0+63 of Bt, contiguous in memory) is
// fetched by the Tensor Data Mover while the waves stage/transform A.
// ---------------------------------------------------------------------------
__global__ __launch_bounds__(256, 2)
void gemm_tile(const float* __restrict__ A0, const float* __restrict__ A1,
               const float* __restrict__ epsp, int epsidx,
               const float* __restrict__ bnA, const float* __restrict__ bnB,
               const __bf16* __restrict__ Bt,
               const float* __restrict__ bias, int bias_n,
               const float* __restrict__ feat1, const float* __restrict__ Wf1,
               const float* __restrict__ feat2, const float* __restrict__ Wf2,
               float* __restrict__ Out, int M,
               float* __restrict__ colsum, float* __restrict__ colsq,
               int amode, int use_small, int use_stats)
{
    __shared__ __align__(16) __bf16 As[TM][KC];   // 20 KB, double-pass over K
    __shared__ __align__(16) __bf16 Bs[TN][HP];   // 40 KB, resident weight panel
    __shared__ float redS[2][TN];

    const int tid  = threadIdx.x;
    const int row0 = blockIdx.x * TM;
    const int col0 = blockIdx.y * TN;

    const float eps1 = (amode == A_PRE) ? (1.0f + epsp[epsidx]) : 1.0f;

    const int lane = tid & 31;
    const int wave = tid >> 5;
    const int wm   = wave >> 1;      // 0..3  -> 16-row band
    const int wn   = wave & 1;       // 0..1  -> 32-col band
    const int m16  = lane & 15;
    const int kh   = lane >> 4;      // K-half selector per ISA 16-bit layout

    // ---- fetch whole B panel: 64 rows x 320 bf16 = 40960 contiguous bytes ----
#if HAVE_TDM
    if (tid < 32) {
        const unsigned long long ga =
            (unsigned long long)(uintptr_t)(Bt + (size_t)col0 * HP);
        const unsigned int lds_off = (unsigned int)(uintptr_t)(&Bs[0][0]);
        v4u g0;
        g0[0] = 1u;                                   // count=1, user descriptor
        g0[1] = lds_off;                              // lds_addr (bytes)
        g0[2] = (unsigned int)ga;                     // global_addr[31:0]
        g0[3] = (unsigned int)(ga >> 32) | 0x80000000u; // addr[56:32] | type=2
        v8i g1;
        g1[0] = 0x00010000;                           // data_size = 2 bytes
        g1[1] = (int)(20480u << 16);                  // tensor_dim0 = 20480 (lo16)
        g1[2] = (int)(1u << 16);                      // dim0 hi=0 | tensor_dim1 = 1
        g1[3] = (int)(20480u << 16);                  // tile_dim0 = 20480
        g1[4] = 1;                                    // tile_dim1 = 1
        g1[5] = 20480;                                // tensor_dim0_stride
        g1[6] = 0;
        g1[7] = 0;
        v4i z4 = {0, 0, 0, 0};                        // groups 2/3 unused (<=2D)
#if __clang_major__ >= 23
        v8i z8 = {0, 0, 0, 0, 0, 0, 0, 0};
        __builtin_amdgcn_tensor_load_to_lds(g0, g1, z4, z4, z8, 0);
#else
        __builtin_amdgcn_tensor_load_to_lds(g0, g1, z4, z4, 0);
#endif
        __builtin_amdgcn_s_wait_tensorcnt(0);         // TENSORcnt is per-wave
    }
#else
    {
        const uint4* gp = (const uint4*)(Bt + (size_t)col0 * HP);
        uint4* lp = (uint4*)(&Bs[0][0]);
        const int r = tid >> 2, c4 = tid & 3;         // 4 threads/row, 40 uint4/row
        for (int kq = c4; kq < HP / 8; kq += 4)
            lp[r * (HP / 8) + kq] = gp[r * (HP / 8) + kq];
    }
#endif

    v8f acc0 = {};
    v8f acc1 = {};

    const int ar  = tid >> 2;        // A staging: 4 threads per row
    const int ac4 = tid & 3;
    const int arow  = wm * 16 + m16;
    const int brow0 = wn * 32 + m16;

    for (int k0 = 0; k0 < HP; k0 += KC) {
        // ---- stage A chunk: load fp32, fuse transform, convert to bf16 ----
        {
            const int row = row0 + ar;
            for (int kq = ac4; kq < KC / 4; kq += 4) {
                const int k = k0 + kq * 4;
                float4 v = make_float4(0.f, 0.f, 0.f, 0.f);
                if (row < M) {
                    const float* ap = A0 + (size_t)row * HP + k;
                    if (k0 == 0) __builtin_prefetch(ap + KC, 0, 1); // next chunk
                    const float4 a = *(const float4*)ap;
                    if (amode == A_PRE) {
                        const float4 g = *(const float4*)(A1 + (size_t)row * HP + k);
                        v.x = eps1 * a.x + g.x; v.y = eps1 * a.y + g.y;
                        v.z = eps1 * a.z + g.z; v.w = eps1 * a.w + g.w;
                    } else if (amode == A_BNRELU) {
                        v.x = fmaxf(bnA[k + 0] * a.x + bnB[k + 0], 0.f);
                        v.y = fmaxf(bnA[k + 1] * a.y + bnB[k + 1], 0.f);
                        v.z = fmaxf(bnA[k + 2] * a.z + bnB[k + 2], 0.f);
                        v.w = fmaxf(bnA[k + 3] * a.w + bnB[k + 3], 0.f);
                    } else {
                        v = a;
                    }
                }
                __bf16* dp = &As[ar][kq * 4];
                dp[0] = (__bf16)v.x; dp[1] = (__bf16)v.y;
                dp[2] = (__bf16)v.z; dp[3] = (__bf16)v.w;
            }
        }
        __syncthreads();   // A chunk ready; also publishes TDM B panel (k0==0)

        // ---- WMMA over this K chunk ----
        for (int kc = 0; kc < KC; kc += 32) {
            union { v16bf v; uint4 q[2]; } av, bv0, bv1;
            const int kb = k0 + kc + kh * 8;
            av.q[0]  = *(const uint4*)(&As[arow][kc + kh * 8]);
            av.q[1]  = *(const uint4*)(&As[arow][kc + kh * 8 + 16]);
            bv0.q[0] = *(const uint4*)(&Bs[brow0][kb]);
            bv0.q[1] = *(const uint4*)(&Bs[brow0][kb + 16]);
            bv1.q[0] = *(const uint4*)(&Bs[brow0 + 16][kb]);
            bv1.q[1] = *(const uint4*)(&Bs[brow0 + 16][kb + 16]);
            acc0 = __builtin_amdgcn_wmma_f32_16x16x32_bf16(
                       false, av.v, false, bv0.v, (short)0, acc0, false, false);
            acc1 = __builtin_amdgcn_wmma_f32_16x16x32_bf16(
                       false, av.v, false, bv1.v, (short)0, acc1, false, false);
        }
        __syncthreads();
    }

    // ---- epilogue: bias (+ fused small-K edge features), store, BN stats ----
    if (use_stats) {
        for (int i = tid; i < 2 * TN; i += 256) (&redS[0][0])[i] = 0.f;
        __syncthreads();
    }

    for (int s = 0; s < 2; ++s) {
        v8f acc;
        if (s == 0) acc = acc0; else acc = acc1;
        const int col  = col0 + wn * 32 + s * 16 + m16;
        const float bv = (col < bias_n) ? bias[col] : 0.f;

        float wf1v[F1N], wf2v[F2N];
        if (use_small) {
            for (int j = 0; j < F1N; ++j) wf1v[j] = Wf1[j * HP + col];
            for (int j = 0; j < F2N; ++j) wf2v[j] = Wf2[j * HP + col];
        }
        float ps = 0.f, pq = 0.f;
        for (int r = 0; r < 8; ++r) {
            const int row = row0 + wm * 16 + kh * 8 + r;
            if (row < M) {
                float v = acc[r] + bv;
                if (use_small) {
                    const float* f1p = feat1 + (size_t)row * F1N;
                    const float* f2p = feat2 + (size_t)row * F2N;
                    for (int j = 0; j < F1N; ++j) v += f1p[j] * wf1v[j];
                    for (int j = 0; j < F2N; ++j) v += f2p[j] * wf2v[j];
                }
                Out[(size_t)row * HP + col] = v;
                ps += v; pq += v * v;
            }
        }
        if (use_stats) {
            atomicAdd(&redS[0][wn * 32 + s * 16 + m16], ps);
            atomicAdd(&redS[1][wn * 32 + s * 16 + m16], pq);
        }
    }
    if (use_stats) {
        __syncthreads();
        for (int i = tid; i < TN; i += 256) {
            atomicAdd(colsum + col0 + i, redS[0][i]);
            atomicAdd(colsq  + col0 + i, redS[1][i]);
        }
    }
}

// ---------------------------------------------------------------------------
// Streaming / prep kernels (bandwidth-bound; width = HP threads per row)
// ---------------------------------------------------------------------------
__global__ void k_zero(float* __restrict__ p, long n)
{
    long i = (long)blockIdx.x * blockDim.x + threadIdx.x;
    const long st = (long)gridDim.x * blockDim.x;
    for (; i < n; i += st) p[i] = 0.f;
}

__global__ void k_copy(const float* __restrict__ s, float* __restrict__ d, long n)
{
    long i = (long)blockIdx.x * blockDim.x + threadIdx.x;
    const long st = (long)gridDim.x * blockDim.x;
    for (; i < n; i += st) d[i] = s[i];
}

__global__ void k_embed_sum(const int* __restrict__ idx, int nid,
                            const float* __restrict__ emb, float* __restrict__ out)
{
    const int r = blockIdx.x, c = threadIdx.x;
    float v = 0.f;
    if (c < HH)
        for (int j = 0; j < nid; ++j)
            v += emb[(size_t)idx[(size_t)r * nid + j] * HH + c];
    out[(size_t)r * HP + c] = v;
}

__global__ void k_prep_wt(const float* __restrict__ W, __bf16* __restrict__ Wt)
{
    const int i = blockIdx.x * blockDim.x + threadIdx.x;
    if (i >= HP * HP) return;
    const int n = i / HP, k = i % HP;
    const float v = (n < HH && k < HH) ? W[(size_t)k * HH + n] : 0.f;
    Wt[i] = (__bf16)v;
}

__global__ void k_prep_fused(const float* __restrict__ linw, int J,
                             const float* __restrict__ Wcat, int rowoff,
                             float* __restrict__ Wf)
{
    const int i = blockIdx.x * blockDim.x + threadIdx.x;
    if (i >= J * HP) return;
    const int j = i / HP, n = i % HP;
    float v = 0.f;
    if (n < HH)
        for (int k = 0; k < HH; ++k)
            v += linw[(size_t)j * HH + k] * Wcat[(size_t)(rowoff + k) * HH + n];
    Wf[i] = v;
}

__global__ void k_prep_btot(const float* __restrict__ catb,
                            const float* __restrict__ f1b, const float* __restrict__ f2b,
                            const float* __restrict__ Wcat, float* __restrict__ btot)
{
    const int n = blockIdx.x * blockDim.x + threadIdx.x;
    if (n >= HP) return;
    float v = 0.f;
    if (n < HH) {
        v = catb[n];
        for (int k = 0; k < HH; ++k) v += f1b[k] * Wcat[(size_t)(300 + k) * HH + n];
        for (int k = 0; k < HH; ++k) v += f2b[k] * Wcat[(size_t)(600 + k) * HH + n];
    }
    btot[n] = v;
}

__global__ void k_bcast_vn(const float* __restrict__ vne, float* __restrict__ vn)
{
    const int g = blockIdx.x, c = threadIdx.x;
    vn[(size_t)g * HP + c] = (c < HH) ? vne[c] : 0.f;
}

__global__ void k_add_vn(float* __restrict__ h, const float* __restrict__ vn,
                         const int* __restrict__ batch)
{
    const int n = blockIdx.x, c = threadIdx.x;
    h[(size_t)n * HP + c] += vn[(size_t)batch[n] * HP + c];
}

__global__ void k_segsum(const float* __restrict__ src, const int* __restrict__ seg,
                         float* __restrict__ dst)
{
    const int n = blockIdx.x, c = threadIdx.x;
    atomicAdd(&dst[(size_t)seg[n] * HP + c], src[(size_t)n * HP + c]);
}

__global__ void k_msg(const float* __restrict__ h, const float* __restrict__ eac,
                      const int* __restrict__ src, const int* __restrict__ dst,
                      float* __restrict__ agg)
{
    const int e = blockIdx.x, c = threadIdx.x;
    float v = h[(size_t)src[e] * HP + c] + eac[(size_t)e * HP + c];
    v = fmaxf(v, 0.f);
    atomicAdd(&agg[(size_t)dst[e] * HP + c], v);
}

__global__ void k_reduce_bn(float* __restrict__ sum, float* __restrict__ sq,
                            const float* __restrict__ g, const float* __restrict__ b,
                            float* __restrict__ bnA, float* __restrict__ bnB, float invM)
{
    const int c = threadIdx.x;
    if (c >= HP) return;
    float a = 0.f, bb = 0.f;
    if (c < HH) {
        const float mean = sum[c] * invM;
        const float var  = sq[c] * invM - mean * mean;
        a  = g[c] * rsqrtf(var + 1e-5f);
        bb = b[c] - mean * a;
    }
    bnA[c] = a; bnB[c] = bb;
    sum[c] = 0.f; sq[c] = 0.f;           // ready for next stats pass
}

__global__ void k_affine_relu(const float* __restrict__ src, const float* __restrict__ bnA,
                              const float* __restrict__ bnB, float* __restrict__ dst)
{
    const long n = blockIdx.x; const int c = threadIdx.x;
    dst[n * HP + c] = fmaxf(bnA[c] * src[n * HP + c] + bnB[c], 0.f);
}

__global__ void k_pred(const float* __restrict__ hg, const float* __restrict__ w,
                       const float* __restrict__ b, float* __restrict__ out)
{
    __shared__ float red[256];
    const int g = blockIdx.x, t = threadIdx.x;
    float s = 0.f;
    for (int c = t; c < HH; c += 256) s += hg[(size_t)g * HP + c] * w[c];
    red[t] = s; __syncthreads();
    for (int k = 128; k > 0; k >>= 1) { if (t < k) red[t] += red[t + k]; __syncthreads(); }
    if (t == 0) out[g] = red[0] + b[0];
}

// ---------------------------------------------------------------------------
extern "C" void kernel_launch(void* const* d_in, const int* in_sizes, int n_in,
                              void* d_out, int out_size, void* d_ws, size_t ws_size,
                              hipStream_t stream)
{
    (void)in_sizes; (void)n_in; (void)out_size; (void)ws_size;

    const int*   x      = (const int*)d_in[0];
    const int*   eai    = (const int*)d_in[1];
    const int*   eidx   = (const int*)d_in[2];
    const int*   batch  = (const int*)d_in[3];
    const float* feat1  = (const float*)d_in[4];
    const float* feat2  = (const float*)d_in[5];
    const float* x_emb  = (const float*)d_in[6];
    const float* e_emb  = (const float*)d_in[7];
    const float* f1w    = (const float*)d_in[8];
    const float* f1b    = (const float*)d_in[9];
    const float* f2w    = (const float*)d_in[10];
    const float* f2b    = (const float*)d_in[11];
    const float* catw   = (const float*)d_in[12];
    const float* catb   = (const float*)d_in[13];
    const float* vne    = (const float*)d_in[14];
    const float* eps    = (const float*)d_in[15];
    const float* W1     = (const float*)d_in[16];
    const float* b1     = (const float*)d_in[17];
    const float* g1     = (const float*)d_in[18];
    const float* be1    = (const float*)d_in[19];
    const float* W2     = (const float*)d_in[20];
    const float* b2     = (const float*)d_in[21];
    const float* bng    = (const float*)d_in[22];
    const float* bnb    = (const float*)d_in[23];
    const float* vW1    = (const float*)d_in[24];
    const float* vb1    = (const float*)d_in[25];
    const float* vg1    = (const float*)d_in[26];
    const float* vbe1   = (const float*)d_in[27];
    const float* vW2    = (const float*)d_in[28];
    const float* vb2    = (const float*)d_in[29];
    const float* vg2    = (const float*)d_in[30];
    const float* vbe2   = (const float*)d_in[31];
    const float* predw  = (const float*)d_in[32];
    const float* predb  = (const float*)d_in[33];
    float* out = (float*)d_out;

    char* base = (char*)d_ws;
    size_t off = 0;
    auto alloc = [&](size_t bytes) -> void* {
        void* p = base + off;
        off += (bytes + 255) & ~(size_t)255;
        return p;
    };

    float* h    = (float*)alloc((size_t)NN * HP * 4);
    float* eac  = (float*)alloc((size_t)EE * HP * 4);
    float* agg  = (float*)alloc((size_t)NN * HP * 4);   // also ea, t2
    float* t1   = (float*)alloc((size_t)NN * HP * 4);
    float* vn   = (float*)alloc((size_t)GG * HP * 4);
    float* vtmp = (float*)alloc((size_t)GG * HP * 4);   // also hg
    float* vt1  = (float*)alloc((size_t)GG * HP * 4);
    float* vt2  = (float*)alloc((size_t)GG * HP * 4);
    __bf16* wt_cat = (__bf16*)alloc((size_t)HP * HP * 2);
    __bf16* wt_W1[LL]; __bf16* wt_W2[LL]; __bf16* wt_V1[LL - 1]; __bf16* wt_V2[LL - 1];
    for (int l = 0; l < LL; ++l)     wt_W1[l] = (__bf16*)alloc((size_t)HP * HP * 2);
    for (int l = 0; l < LL; ++l)     wt_W2[l] = (__bf16*)alloc((size_t)HP * HP * 2);
    for (int l = 0; l < LL - 1; ++l) wt_V1[l] = (__bf16*)alloc((size_t)HP * HP * 2);
    for (int l = 0; l < LL - 1; ++l) wt_V2[l] = (__bf16*)alloc((size_t)HP * HP * 2);
    float* Wf1  = (float*)alloc((size_t)F1N * HP * 4);
    float* Wf2  = (float*)alloc((size_t)F2N * HP * 4);
    float* btot = (float*)alloc((size_t)HP * 4);
    float* ssum = (float*)alloc((size_t)HP * 4);
    float* ssq  = (float*)alloc((size_t)HP * 4);
    float* bnA  = (float*)alloc((size_t)HP * 4);
    float* bnB  = (float*)alloc((size_t)HP * 4);

    const dim3 wblk(256), wgrd((HP * HP + 255) / 256);

    // ---- weight prep: transpose + pad + bf16; fuse f1/f2 into cat ----
    k_prep_wt<<<wgrd, wblk, 0, stream>>>(catw, wt_cat);
    for (int l = 0; l < LL; ++l)
        k_prep_wt<<<wgrd, wblk, 0, stream>>>(W1 + (size_t)l * HH * HH, wt_W1[l]);
    for (int l = 0; l < LL; ++l)
        k_prep_wt<<<wgrd, wblk, 0, stream>>>(W2 + (size_t)l * HH * HH, wt_W2[l]);
    for (int l = 0; l < LL - 1; ++l)
        k_prep_wt<<<wgrd, wblk, 0, stream>>>(vW1 + (size_t)l * HH * HH, wt_V1[l]);
    for (int l = 0; l < LL - 1; ++l)
        k_prep_wt<<<wgrd, wblk, 0, stream>>>(vW2 + (size_t)l * HH * HH, wt_V2[l]);
    k_prep_fused<<<(F1N * HP + 255) / 256, 256, 0, stream>>>(f1w, F1N, catw, 300, Wf1);
    k_prep_fused<<<(F2N * HP + 255) / 256, 256, 0, stream>>>(f2w, F2N, catw, 600, Wf2);
    k_prep_btot<<<(HP + 255) / 256, 256, 0, stream>>>(catb, f1b, f2b, catw, btot);
    k_zero<<<2, 256, 0, stream>>>(ssum, HP);
    k_zero<<<2, 256, 0, stream>>>(ssq, HP);

    // ---- embeddings ----
    k_embed_sum<<<NN, HP, 0, stream>>>(x, 9, x_emb, h);
    k_embed_sum<<<EE, HP, 0, stream>>>(eai, 3, e_emb, agg);   // ea -> agg

    // ---- edge features: ea @ Wcat_top + feat1@Wf1 + feat2@Wf2 + btot ----
    const dim3 gblk(256);
    const dim3 ggE((EE + TM - 1) / TM, HP / TN);
    const dim3 ggN((NN + TM - 1) / TM, HP / TN);
    const dim3 ggG((GG + TM - 1) / TM, HP / TN);
    gemm_tile<<<ggE, gblk, 0, stream>>>(agg, nullptr, nullptr, 0, nullptr, nullptr,
        wt_cat, btot, HP, feat1, Wf1, feat2, Wf2, eac, EE, nullptr, nullptr,
        A_PLAIN, 1, 0);

    k_bcast_vn<<<GG, HP, 0, stream>>>(vne, vn);
    const float invN = 1.0f / NN, invG = 1.0f / GG;

    for (int l = 0; l < LL; ++l) {
        k_add_vn<<<NN, HP, 0, stream>>>(h, vn, batch);
        if (l < LL - 1) {
            k_copy<<<2048, 256, 0, stream>>>(vn, vtmp, (long)GG * HP);
            k_segsum<<<NN, HP, 0, stream>>>(h, batch, vtmp);
        }
        k_zero<<<4096, 256, 0, stream>>>(agg, (long)NN * HP);
        k_msg<<<EE, HP, 0, stream>>>(h, eac, eidx, eidx + EE, agg);

        // GIN MLP: gemm1 (pre = (1+eps)h + agg) -> BN stats; reduce; gemm2
        gemm_tile<<<ggN, gblk, 0, stream>>>(h, agg, eps, l, nullptr, nullptr,
            wt_W1[l], b1 + (size_t)l * HH, HH, nullptr, nullptr, nullptr, nullptr,
            t1, NN, ssum, ssq, A_PRE, 0, 1);
        k_reduce_bn<<<1, HP, 0, stream>>>(ssum, ssq, g1 + (size_t)l * HH,
                                          be1 + (size_t)l * HH, bnA, bnB, invN);
        gemm_tile<<<ggN, gblk, 0, stream>>>(t1, nullptr, nullptr, 0, bnA, bnB,
            wt_W2[l], b2 + (size_t)l * HH, HH, nullptr, nullptr, nullptr, nullptr,
            agg /* t2 */, NN, ssum, ssq, A_BNRELU, 0, 1);
        k_reduce_bn<<<1, HP, 0, stream>>>(ssum, ssq, bng + (size_t)l * HH,
                                          bnb + (size_t)l * HH, bnA, bnB, invN);
        k_affine_relu<<<NN, HP, 0, stream>>>(agg, bnA, bnB, h);

        if (l < LL - 1) {   // virtual node MLP on vtmp (G rows)
            gemm_tile<<<ggG, gblk, 0, stream>>>(vtmp, nullptr, nullptr, 0, nullptr, nullptr,
                wt_V1[l], vb1 + (size_t)l * HH, HH, nullptr, nullptr, nullptr, nullptr,
                vt1, GG, ssum, ssq, A_PLAIN, 0, 1);
            k_reduce_bn<<<1, HP, 0, stream>>>(ssum, ssq, vg1 + (size_t)l * HH,
                                              vbe1 + (size_t)l * HH, bnA, bnB, invG);
            gemm_tile<<<ggG, gblk, 0, stream>>>(vt1, nullptr, nullptr, 0, bnA, bnB,
                wt_V2[l], vb2 + (size_t)l * HH, HH, nullptr, nullptr, nullptr, nullptr,
                vt2, GG, ssum, ssq, A_BNRELU, 0, 1);
            k_reduce_bn<<<1, HP, 0, stream>>>(ssum, ssq, vg2 + (size_t)l * HH,
                                              vbe2 + (size_t)l * HH, bnA, bnB, invG);
            k_affine_relu<<<GG, HP, 0, stream>>>(vt2, bnA, bnB, vn);
        }
    }

    // ---- readout ----
    k_zero<<<2048, 256, 0, stream>>>(vtmp, (long)GG * HP);
    k_segsum<<<NN, HP, 0, stream>>>(h, batch, vtmp);
    k_pred<<<GG, 256, 0, stream>>>(vtmp, predw, predb, out);
}
